// GIN_5282809775005
// MI455X (gfx1250) — matrix operand order repaired and checked by reading, not compiled, not run on hardware
//
#include <hip/hip_runtime.h>
#include <math.h>

typedef __bf16 bf16;
typedef __attribute__((ext_vector_type(16))) __bf16 bf16x16;
typedef __attribute__((ext_vector_type(8)))  float  f32x8;
typedef __attribute__((ext_vector_type(4)))  unsigned int u32x4;

#define LDST 144   // bf16 elements per LDS row (128 + pad); 288B row keeps 16B/32B alignment

// ---- fragment helpers (CDNA5 wave32 WMMA layouts, CDNA5 ISA) ----

__device__ __forceinline__ bf16x16 pack_frag(u32x4 lo, u32x4 hi) {
    union { struct { u32x4 a, b; } u; bf16x16 v; } c;
    c.u.a = lo; c.u.b = hi;
    return c.v;
}

// B matrix (KxN, bf16): lane n = lane&15; elems 0..7 = K[k0+half*8 ..], elems 8..15 = K[k0+16+half*8 ..].
// LDS holds W transposed: [n][k], so the 16 K values are contiguous.
__device__ __forceinline__ bf16x16 frag_b_lds(const bf16* base, int row, int k0, int half) {
    const bf16* p = base + row * LDST + k0 + half * 8;
    return pack_frag(*(const u32x4*)p, *(const u32x4*)(p + 16));
}

// A matrix (MxK, bf16) from LDS: lane row = lane&15; elems 0..7 = K[k0+half*8 ..],
// elems 8..15 = K[k0+16+half*8 ..].
__device__ __forceinline__ bf16x16 frag_a_lds(const bf16* base, int row, int k0, int half) {
    const bf16* p = base + row * LDST + k0 + half * 8;
    return pack_frag(*(const u32x4*)p, *(const u32x4*)(p + 16));
}

// A matrix fragment straight from global f32 (row-major, ld=128) with on-the-fly bf16 convert.
__device__ __forceinline__ bf16x16 frag_a_global(const float* __restrict__ X, long row, int k0, int half) {
    const float* p = X + row * 128 + k0 + half * 8;
    bf16x16 a;
#pragma unroll
    for (int i = 0; i < 8; ++i) a[i] = (bf16)p[i];
#pragma unroll
    for (int i = 0; i < 8; ++i) a[8 + i] = (bf16)p[16 + i];
    return a;
}

__device__ __forceinline__ f32x8 wmma_bf16(bf16x16 a, bf16x16 b, f32x8 c) {
    return __builtin_amdgcn_wmma_f32_16x16x32_bf16(false, a, false, b, (short)0, c, false, false);
}

// ---- prep: fold BN into bias/scale ----
__global__ __launch_bounds__(256) void prep_bn_kernel(
    const float* __restrict__ b1, const float* __restrict__ gamma,
    const float* __restrict__ beta, const float* __restrict__ mean,
    const float* __restrict__ var, float* __restrict__ scale,
    float* __restrict__ biaseff, int total)
{
    int i = blockIdx.x * 256 + threadIdx.x;
    if (i < total) {
        float s = gamma[i] * rsqrtf(var[i] + 1e-5f);
        scale[i] = s;
        biaseff[i] = (b1[i] - mean[i]) * s + beta[i];
    }
}

// ---- prep: transpose [K][Nsrc] f32 -> [Npad][K] bf16, optional per-column scale ----
__global__ __launch_bounds__(256) void prep_weight_kernel(
    const float* __restrict__ W, bf16* __restrict__ Wt,
    const float* __restrict__ colscale, int K, int Nsrc, int Npad)
{
    int idx = blockIdx.x * 256 + threadIdx.x;
    if (idx >= Npad * K) return;
    int n = idx % Npad, k = idx / Npad;
    float v = 0.0f;
    if (n < Nsrc) {
        v = W[(size_t)k * Nsrc + n];
        if (colscale) v *= colscale[n];
    }
    Wt[(size_t)n * K + k] = (bf16)v;
}

// ---- scatter: agg[dst] += h[src]; one wave per edge, float4 per lane, hw f32 atomics ----
__global__ __launch_bounds__(256) void gin_scatter_kernel(
    const float* __restrict__ h, const int* __restrict__ ei,
    float* __restrict__ agg, int E)
{
    int idx = blockIdx.x * 256 + threadIdx.x;
    int e = idx >> 5;
    if (e >= E) return;
    int lane = idx & 31;
    int s = ei[e];
    int d = ei[E + e];
    float4 v = ((const float4*)(h + (size_t)s * 128))[lane];
    float* dp = agg + (size_t)d * 128 + (size_t)lane * 4;
    unsafeAtomicAdd(dp + 0, v.x);
    unsafeAtomicAdd(dp + 1, v.y);
    unsafeAtomicAdd(dp + 2, v.z);
    unsafeAtomicAdd(dp + 3, v.w);
}

// ---- fused GIN MLP: H = relu(relu((X @ W1f + b1f)) @ W2 + b2), BN pre-folded into W1f/b1f
__global__ __launch_bounds__(256) void gin_mlp_kernel(
    const float* __restrict__ X,        // [N,128] f32 (agg)
    const bf16*  __restrict__ W1t,      // [128][128] bf16, transposed + BN-scaled
    const float* __restrict__ bias1,    // [128] folded bias
    const bf16*  __restrict__ W2t,      // [128][128] bf16, transposed
    const float* __restrict__ bias2,    // [128]
    float* __restrict__ Hout,           // [N,128] f32
    int N)
{
    __shared__ __align__(32) bf16 Bs[128 * LDST];
    __shared__ __align__(32) bf16 Ts[128 * LDST];
    const f32x8 fzero = {0, 0, 0, 0, 0, 0, 0, 0};

    const int tid = threadIdx.x;
    const int wave = tid >> 5, lane = tid & 31;
    const int half = lane >> 4, l15 = lane & 15;
    const long row0 = (long)blockIdx.x * 128;
    const int wrow = wave * 16;

    // stage W1t (128x128 bf16) into LDS, 16B chunks
    for (int i = tid; i < 2048; i += 256) {
        int r = i >> 4, c = (i & 15) * 8;
        *(u32x4*)&Bs[r * LDST + c] = ((const u32x4*)W1t)[i];
    }
    __syncthreads();

    long arow = row0 + wrow + l15;
    if (arow >= N) arow = N - 1;                 // clamp; garbage rows never stored

    f32x8 acc[8];
#pragma unroll
    for (int n = 0; n < 8; ++n) acc[n] = fzero;

#pragma unroll
    for (int k0 = 0; k0 < 128; k0 += 32) {
        bf16x16 a = frag_a_global(X, arow, k0, half);
#pragma unroll
        for (int n = 0; n < 8; ++n) {
            bf16x16 b = frag_b_lds(Bs, n * 16 + l15, k0, half);
            acc[n] = wmma_bf16(a, b, acc[n]);
        }
    }

    // epilogue 1: folded-BN bias + relu -> bf16 intermediate in LDS
#pragma unroll
    for (int n = 0; n < 8; ++n) {
        int col = n * 16 + l15;
        float bv = bias1[col];
#pragma unroll
        for (int r = 0; r < 8; ++r) {
            float v = fmaxf(acc[n][r] + bv, 0.0f);
            Ts[(wrow + half * 8 + r) * LDST + col] = (bf16)v;
        }
    }
    __syncthreads();

    // stage W2t
    for (int i = tid; i < 2048; i += 256) {
        int r = i >> 4, c = (i & 15) * 8;
        *(u32x4*)&Bs[r * LDST + c] = ((const u32x4*)W2t)[i];
    }
    __syncthreads();

#pragma unroll
    for (int n = 0; n < 8; ++n) acc[n] = fzero;

#pragma unroll
    for (int k0 = 0; k0 < 128; k0 += 32) {
        bf16x16 a = frag_a_lds(Ts, wrow + l15, k0, half);
#pragma unroll
        for (int n = 0; n < 8; ++n) {
            bf16x16 b = frag_b_lds(Bs, n * 16 + l15, k0, half);
            acc[n] = wmma_bf16(a, b, acc[n]);
        }
    }

    // epilogue 2: bias + relu -> f32 global
#pragma unroll
    for (int n = 0; n < 8; ++n) {
        int col = n * 16 + l15;
        float bv = bias2[col];
#pragma unroll
        for (int r = 0; r < 8; ++r) {
            long row = row0 + wrow + half * 8 + r;
            if (row < N) Hout[row * 128 + col] = fmaxf(acc[n][r] + bv, 0.0f);
        }
    }
}

// ---- head: log_softmax(relu(H @ lw1 + lb1) @ lw2 + lb2) ----
__global__ __launch_bounds__(256) void gin_head_kernel(
    const float* __restrict__ X,      // [N,128]
    const bf16*  __restrict__ lw1t,   // [128][128]
    const float* __restrict__ lb1,    // [128]
    const bf16*  __restrict__ lw2t,   // [48][128] (row 47 zero-padded)
    const float* __restrict__ lb2,    // [47]
    float* __restrict__ out,          // [N,47]
    int N)
{
    __shared__ __align__(32) bf16 Bs[128 * LDST];
    __shared__ __align__(32) bf16 Ts[128 * LDST];
    __shared__ float Ls[128 * 52];
    const f32x8 fzero = {0, 0, 0, 0, 0, 0, 0, 0};

    const int tid = threadIdx.x;
    const int wave = tid >> 5, lane = tid & 31;
    const int half = lane >> 4, l15 = lane & 15;
    const long row0 = (long)blockIdx.x * 128;
    const int wrow = wave * 16;

    for (int i = tid; i < 2048; i += 256) {
        int r = i >> 4, c = (i & 15) * 8;
        *(u32x4*)&Bs[r * LDST + c] = ((const u32x4*)lw1t)[i];
    }
    __syncthreads();

    long arow = row0 + wrow + l15;
    if (arow >= N) arow = N - 1;

    f32x8 acc[8];
#pragma unroll
    for (int n = 0; n < 8; ++n) acc[n] = fzero;

#pragma unroll
    for (int k0 = 0; k0 < 128; k0 += 32) {
        bf16x16 a = frag_a_global(X, arow, k0, half);
#pragma unroll
        for (int n = 0; n < 8; ++n) {
            bf16x16 b = frag_b_lds(Bs, n * 16 + l15, k0, half);
            acc[n] = wmma_bf16(a, b, acc[n]);
        }
    }
#pragma unroll
    for (int n = 0; n < 8; ++n) {
        int col = n * 16 + l15;
        float bv = lb1[col];
#pragma unroll
        for (int r = 0; r < 8; ++r) {
            float v = fmaxf(acc[n][r] + bv, 0.0f);
            Ts[(wrow + half * 8 + r) * LDST + col] = (bf16)v;
        }
    }
    __syncthreads();

    // stage lw2t: 48 rows x 128 K = 768 16B chunks
    for (int i = tid; i < 768; i += 256) {
        int r = i >> 4, c = (i & 15) * 8;
        *(u32x4*)&Bs[r * LDST + c] = ((const u32x4*)lw2t)[i];
    }
    __syncthreads();

    f32x8 acc2[3];
#pragma unroll
    for (int n = 0; n < 3; ++n) acc2[n] = fzero;

#pragma unroll
    for (int k0 = 0; k0 < 128; k0 += 32) {
        bf16x16 a = frag_a_lds(Ts, wrow + l15, k0, half);
#pragma unroll
        for (int n = 0; n < 3; ++n) {
            bf16x16 b = frag_b_lds(Bs, n * 16 + l15, k0, half);
            acc2[n] = wmma_bf16(a, b, acc2[n]);
        }
    }
#pragma unroll
    for (int n = 0; n < 3; ++n) {
        int col = n * 16 + l15;
        float bv = (col < 47) ? lb2[col] : 0.0f;
#pragma unroll
        for (int r = 0; r < 8; ++r)
            Ls[(wrow + half * 8 + r) * 52 + col] = acc2[n][r] + bv;
    }
    __syncthreads();

    // per-row log-softmax over 47 classes
    if (tid < 128) {
        long row = row0 + tid;
        if (row < N) {
            const float* lr = &Ls[tid * 52];
            float m = -INFINITY;
#pragma unroll
            for (int c = 0; c < 47; ++c) m = fmaxf(m, lr[c]);
            float s = 0.0f;
#pragma unroll
            for (int c = 0; c < 47; ++c) s += __expf(lr[c] - m);
            float lg = __logf(s) + m;
            float* o = out + row * 47;
#pragma unroll
            for (int c = 0; c < 47; ++c) o[c] = lr[c] - lg;
        }
    }
}

extern "C" void kernel_launch(void* const* d_in, const int* in_sizes, int n_in,
                              void* d_out, int out_size, void* d_ws, size_t ws_size,
                              hipStream_t stream)
{
    const float* x     = (const float*)d_in[0];
    const int*   ei    = (const int*)  d_in[1];
    const float* W1    = (const float*)d_in[2];
    const float* b1    = (const float*)d_in[3];
    const float* gamma = (const float*)d_in[4];
    const float* beta  = (const float*)d_in[5];
    const float* mean  = (const float*)d_in[6];
    const float* var   = (const float*)d_in[7];
    const float* W2    = (const float*)d_in[8];
    const float* b2    = (const float*)d_in[9];
    const float* lw1   = (const float*)d_in[10];
    const float* lb1   = (const float*)d_in[11];
    const float* lw2   = (const float*)d_in[12];
    const float* lb2   = (const float*)d_in[13];
    float* out = (float*)d_out;

    const int Hd = 128, Cd = 47, Ln = 3;
    const int N = in_sizes[0] / Hd;
    const int E = in_sizes[1] / 2;

    char* p = (char*)d_ws;
    float* agg   = (float*)p; p += (size_t)N * Hd * 4;
    float* hbuf  = (float*)p; p += (size_t)N * Hd * 4;
    bf16*  w1t   = (bf16*)p;  p += (size_t)Ln * Hd * Hd * 2;
    bf16*  w2t   = (bf16*)p;  p += (size_t)Ln * Hd * Hd * 2;
    bf16*  lw1t  = (bf16*)p;  p += (size_t)Hd * Hd * 2;
    bf16*  lw2t  = (bf16*)p;  p += (size_t)48 * Hd * 2;
    float* scale = (float*)p; p += (size_t)Ln * Hd * 4;
    float* b1e   = (float*)p; p += (size_t)Ln * Hd * 4;

    // weight prep (tiny)
    prep_bn_kernel<<<(Ln * Hd + 255) / 256, 256, 0, stream>>>(b1, gamma, beta, mean, var, scale, b1e, Ln * Hd);
    const int wblk = (Hd * Hd + 255) / 256;
    for (int l = 0; l < Ln; ++l) {
        prep_weight_kernel<<<wblk, 256, 0, stream>>>(W1 + (size_t)l * Hd * Hd, w1t + (size_t)l * Hd * Hd,
                                                     scale + l * Hd, Hd, Hd, Hd);
        prep_weight_kernel<<<wblk, 256, 0, stream>>>(W2 + (size_t)l * Hd * Hd, w2t + (size_t)l * Hd * Hd,
                                                     nullptr, Hd, Hd, Hd);
    }
    prep_weight_kernel<<<wblk, 256, 0, stream>>>(lw1, lw1t, nullptr, Hd, Hd, Hd);
    prep_weight_kernel<<<(48 * Hd + 255) / 256, 256, 0, stream>>>(lw2, lw2t, nullptr, Hd, Cd, 48);

    const int nblk = (N + 127) / 128;
    const long sth = (long)E * 32;
    const int sblk = (int)((sth + 255) / 256);

    const float* hcur = x;
    for (int l = 0; l < Ln; ++l) {
        // agg = h  (the (1+eps)*x term, eps=0), then += sum over neighbors
        hipMemcpyAsync(agg, hcur, (size_t)N * Hd * 4, hipMemcpyDeviceToDevice, stream);
        gin_scatter_kernel<<<sblk, 256, 0, stream>>>(hcur, ei, agg, E);
        gin_mlp_kernel<<<nblk, 256, 0, stream>>>(agg, w1t + (size_t)l * Hd * Hd, b1e + l * Hd,
                                                 w2t + (size_t)l * Hd * Hd, b2 + (size_t)l * Hd, hbuf, N);
        hcur = hbuf;
    }
    gin_head_kernel<<<nblk, 256, 0, stream>>>(hcur, lw1t, lb1, lw2t, lb2, out, N);
}
